// MultiheadAttention_73315091742893
// MI455X (gfx1250) — matrix operand order repaired and checked by reading, not compiled, not run on hardware
//
#include <hip/hip_runtime.h>
#include <hip/hip_bf16.h>

// ---------------------------------------------------------------------------
// MultiheadAttention forward for MI455X (gfx1250), bf16 WMMA path.
//   d_out = [ out (2*2048*1024 f32) | attn (2*16*2048*2048 f32) ]
// ---------------------------------------------------------------------------

typedef __bf16 v16bf __attribute__((ext_vector_type(16)));
typedef __bf16 v8bf  __attribute__((ext_vector_type(8)));
typedef __bf16 v4bf  __attribute__((ext_vector_type(4)));
typedef float  v8f   __attribute__((ext_vector_type(8)));
typedef float  v4f   __attribute__((ext_vector_type(4)));

#define D_MODEL 1024
#define N_HEADS 16
#define D_K     64
#define SEQ     2048
#define BATCH   2
#define M_TOT   (BATCH * SEQ)   // 4096

#define LDS_STRIDE 40           // padded [n][k] tile stride (16B aligned, bank-spread)

// float -> bf16, round-to-nearest-even
__device__ __forceinline__ __bf16 f2bf(float f) {
    union { float f; unsigned int u; } x; x.f = f;
    unsigned int r = x.u + 0x7FFFu + ((x.u >> 16) & 1u);
    union { unsigned short u; __bf16 b; } y; y.u = (unsigned short)(r >> 16);
    return y.b;
}

// ---------------------------------------------------------------------------
// elementwise f32 -> bf16 (vectorized: b128 load / b64 store)
// ---------------------------------------------------------------------------
__global__ __launch_bounds__(256) void cvt_f32_bf16(const float* __restrict__ src,
                                                    __bf16* __restrict__ dst, int n4) {
    int i = blockIdx.x * 256 + threadIdx.x;
    if (i < n4) {
        v4f s = *(const v4f*)(src + (size_t)i * 4);
        v4bf d;
#pragma unroll
        for (int e = 0; e < 4; ++e) d[e] = f2bf(s[e]);
        *(v4bf*)(dst + (size_t)i * 4) = d;
    }
}

// ---------------------------------------------------------------------------
// A-fragment loaders (ISA 16-bit A 16x32 layout): two aligned chunks of 8
// elements at +0 and +16; ap already offset by arow*lda + k0 + (lane>=16)*8.
// ---------------------------------------------------------------------------
__device__ __forceinline__ v16bf load_afrag(const __bf16* ap) {
    v8bf lo = *(const v8bf*)ap;          // global_load_b128
    v8bf hi = *(const v8bf*)(ap + 16);   // global_load_b128
    v16bf r;
#pragma unroll
    for (int e = 0; e < 8; ++e) { r[e] = lo[e]; r[8 + e] = hi[e]; }
    return r;
}
__device__ __forceinline__ v16bf load_afrag(const float* ap) {
    v4f c0 = *(const v4f*)ap,        c1 = *(const v4f*)(ap + 4);
    v4f c2 = *(const v4f*)(ap + 16), c3 = *(const v4f*)(ap + 20);
    v16bf r;
#pragma unroll
    for (int e = 0; e < 4; ++e) {
        r[e]      = f2bf(c0[e]);  r[4 + e]  = f2bf(c1[e]);
        r[8 + e]  = f2bf(c2[e]);  r[12 + e] = f2bf(c3[e]);
    }
    return r;
}

// ---------------------------------------------------------------------------
// Generic bf16 WMMA GEMM:  C[z] = scale * (A[z] @ op(B[z])) (+ bias)
//   AT : float (converted on the fly) or __bf16
//   BT : true  -> B is [N,K] row-major (A @ B^T); fragments are two 16B chunks
//                 32B apart per lane, loaded directly with 2x global_load_b128.
//        false -> B is [K,N] row-major; K32xN64 tile staged in LDS as a
//                 padded transposed [n][k] tile: one global_load_b128 per
//                 thread + 8 ds_store_b16, fragments read as 2x ds_load_b128.
//   Block: 256 threads = 8 waves; macro-tile 256(M) x 64(N); BLK_K = 32.
//   Each wave: 32x64 strip = 2 M-subtiles x 4 N-subtiles = 8 WMMA / K-step,
//   B fragments reused across M-subtiles, A fragments across N-subtiles.
// ---------------------------------------------------------------------------
template <typename AT, bool BT, bool BIAS, bool OUTBF16>
__global__ __launch_bounds__(256) void wmma_gemm(
    const AT* __restrict__ Abase, const __bf16* __restrict__ Bbase,
    const float* __restrict__ bias, void* __restrict__ Cbase,
    int K, int lda, int ldb, int ldc, int zH,
    long long sAb, long long sAh, long long sBb, long long sBh,
    long long sCb, long long sCh, float scale)
{
    __shared__ __bf16 bsh[64 * LDS_STRIDE];   // 5 KB

    const int tid  = threadIdx.x;
    const int wave = tid >> 5;
    const int lane = tid & 31;
    const int zb   = blockIdx.z / zH;
    const int zh   = blockIdx.z % zH;

    const AT*     A = Abase + (size_t)zb * sAb + (size_t)zh * sAh;
    const __bf16* B = Bbase + (size_t)zb * sBb + (size_t)zh * sBh;

    const int rowBase = blockIdx.y * 256 + wave * 32;
    const int colBase = blockIdx.x * 64;

    const int arow  = rowBase + (lane & 15);
    const int aoff  = (lane >> 4) * 8;       // 0: K 0..7/16..23, 1: K 8..15/24..31
    const int kbase = (lane >> 4) << 4;      // unused (B fragments use aoff, 0 or 8)

    // staging map: thread -> (k row, 8-wide n chunk) of the K32 x N64 B tile
    const int skk  = tid >> 3;               // 0..31
    const int snch = (tid & 7) * 8;          // 0..56

    v8f acc[2][4];
#pragma unroll
    for (int m = 0; m < 2; ++m)
#pragma unroll
        for (int t = 0; t < 4; ++t) acc[m][t] = (v8f)(0.0f);

    for (int k0 = 0; k0 < K; k0 += 32) {
        // ---- A fragments (2 M-subtiles) --------------------------------
        const AT* ap0 = A + (size_t)arow * lda + k0 + aoff;
        const AT* ap1 = ap0 + (size_t)16 * lda;
        v16bf af0 = load_afrag(ap0);
        v16bf af1 = load_afrag(ap1);
        __builtin_prefetch(ap0 + 32, 0, 1);   // next K tile -> global_prefetch_b8
        __builtin_prefetch(ap1 + 32, 0, 1);

        if (!BT) {
            __syncthreads();
            const __bf16* bp = B + (size_t)(k0 + skk) * ldb + colBase + snch;
            v8bf chunk = *(const v8bf*)bp;            // global_load_b128
            __builtin_prefetch(bp + (size_t)32 * ldb, 0, 1);
#pragma unroll
            for (int j = 0; j < 8; ++j)
                bsh[(snch + j) * LDS_STRIDE + skk] = chunk[j];
            __syncthreads();
        }

#pragma unroll
        for (int t = 0; t < 4; ++t) {
            v16bf bfrag;
            if (BT) {
                // B row-major [N,K]: lane holds col = lane&15,
                // K = k0 + aoff + {0..7, 16..23} -> two 8-elem chunks (2x b128)
                const __bf16* bp = B + (size_t)(colBase + t * 16 + (lane & 15)) * ldb
                                     + k0 + aoff;
                v8bf lo = *(const v8bf*)bp;
                v8bf hi = *(const v8bf*)(bp + 16);
#pragma unroll
                for (int e = 0; e < 8; ++e) { bfrag[e] = lo[e]; bfrag[8 + e] = hi[e]; }
            } else {
                const __bf16* p = &bsh[(t * 16 + (lane & 15)) * LDS_STRIDE + aoff];
                v8bf lo = *(const v8bf*)p;            // ds_load_b128
                v8bf hi = *(const v8bf*)(p + 16);     // ds_load_b128
#pragma unroll
                for (int e = 0; e < 8; ++e) { bfrag[e] = lo[e]; bfrag[8 + e] = hi[e]; }
            }
            acc[0][t] = __builtin_amdgcn_wmma_f32_16x16x32_bf16(
                false, af0, false, bfrag, (short)0, acc[0][t], false, false);
            acc[1][t] = __builtin_amdgcn_wmma_f32_16x16x32_bf16(
                false, af1, false, bfrag, (short)0, acc[1][t], false, false);
        }
    }

    // ---- epilogue: C/D layout — VGPR v: row = v + 8*(lane>=16), col = lane&15
    const size_t cz = (size_t)zb * sCb + (size_t)zh * sCh;
#pragma unroll
    for (int m = 0; m < 2; ++m) {
#pragma unroll
        for (int t = 0; t < 4; ++t) {
            int col = colBase + t * 16 + (lane & 15);
            float bv = BIAS ? bias[col] : 0.0f;
#pragma unroll
            for (int v = 0; v < 8; ++v) {
                int row = rowBase + m * 16 + v + ((lane >> 4) << 3);
                float val = acc[m][t][v] * scale + bv;
                size_t off = cz + (size_t)row * ldc + col;
                if (OUTBF16) ((__bf16*)Cbase)[off] = f2bf(val);
                else         ((float*)Cbase)[off]  = val;
            }
        }
    }
}

// ---------------------------------------------------------------------------
// In-place softmax over rows of length 2048 (one 256-thread block per row)
// ---------------------------------------------------------------------------
__global__ __launch_bounds__(256) void softmax_rows2048(float* __restrict__ attn) {
    __shared__ float red[256];
    float* row = attn + (size_t)blockIdx.x * 2048;
    const int tid = threadIdx.x;

    float vals[8];
    float m = -1e30f;
#pragma unroll
    for (int j = 0; j < 8; ++j) {
        vals[j] = row[tid + j * 256];
        m = fmaxf(m, vals[j]);
    }
    red[tid] = m; __syncthreads();
#pragma unroll
    for (int s = 128; s > 0; s >>= 1) {
        if (tid < s) red[tid] = fmaxf(red[tid], red[tid + s]);
        __syncthreads();
    }
    m = red[0]; __syncthreads();

    float sum = 0.0f;
#pragma unroll
    for (int j = 0; j < 8; ++j) {
        vals[j] = __expf(vals[j] - m);
        sum += vals[j];
    }
    red[tid] = sum; __syncthreads();
#pragma unroll
    for (int s = 128; s > 0; s >>= 1) {
        if (tid < s) red[tid] += red[tid + s];
        __syncthreads();
    }
    float inv = 1.0f / red[0];
#pragma unroll
    for (int j = 0; j < 8; ++j) row[tid + j * 256] = vals[j] * inv;
}

// ---------------------------------------------------------------------------
extern "C" void kernel_launch(void* const* d_in, const int* in_sizes, int n_in,
                              void* d_out, int out_size, void* d_ws, size_t ws_size,
                              hipStream_t stream) {
    (void)in_sizes; (void)n_in; (void)out_size; (void)ws_size;

    const float* X  = (const float*)d_in[0];
    const float* Wq = (const float*)d_in[1];
    const float* bq = (const float*)d_in[2];
    const float* Wk = (const float*)d_in[3];
    const float* bk = (const float*)d_in[4];
    const float* Wv = (const float*)d_in[5];
    const float* bv = (const float*)d_in[6];
    const float* Wo = (const float*)d_in[7];
    const float* bo = (const float*)d_in[8];

    float* out  = (float*)d_out;
    float* attn = out + (size_t)M_TOT * D_MODEL;   // 2*16*2048*2048 f32

    // workspace (bf16): q,k,v,ctx,Xb (4M each) + 4 weights (1M each) = 48 MB
    __bf16* qb  = (__bf16*)d_ws;
    __bf16* kb  = qb  + (size_t)M_TOT * D_MODEL;
    __bf16* vb  = kb  + (size_t)M_TOT * D_MODEL;
    __bf16* ctx = vb  + (size_t)M_TOT * D_MODEL;
    __bf16* Xb  = ctx + (size_t)M_TOT * D_MODEL;
    __bf16* Wqb = Xb  + (size_t)M_TOT * D_MODEL;
    __bf16* Wkb = Wqb + (size_t)D_MODEL * D_MODEL;
    __bf16* Wvb = Wkb + (size_t)D_MODEL * D_MODEL;
    __bf16* Wob = Wvb + (size_t)D_MODEL * D_MODEL;

    const int nX4 = (M_TOT * D_MODEL) / 4;       // 1048576
    const int nW4 = (D_MODEL * D_MODEL) / 4;     // 262144
    cvt_f32_bf16<<<nX4 / 256, 256, 0, stream>>>(X,  Xb,  nX4);
    cvt_f32_bf16<<<nW4 / 256, 256, 0, stream>>>(Wq, Wqb, nW4);
    cvt_f32_bf16<<<nW4 / 256, 256, 0, stream>>>(Wk, Wkb, nW4);
    cvt_f32_bf16<<<nW4 / 256, 256, 0, stream>>>(Wv, Wvb, nW4);
    cvt_f32_bf16<<<nW4 / 256, 256, 0, stream>>>(Wo, Wob, nW4);

    // -------- QKV projections: [4096,1024] = Xb @ W + b  (bf16 out) --------
    dim3 gProj(D_MODEL / 64, M_TOT / 256, 1);
    wmma_gemm<__bf16, false, true, true><<<gProj, 256, 0, stream>>>(
        Xb, Wqb, bq, qb, D_MODEL, D_MODEL, D_MODEL, D_MODEL,
        1, 0, 0, 0, 0, 0, 0, 1.0f);
    wmma_gemm<__bf16, false, true, true><<<gProj, 256, 0, stream>>>(
        Xb, Wkb, bk, kb, D_MODEL, D_MODEL, D_MODEL, D_MODEL,
        1, 0, 0, 0, 0, 0, 0, 1.0f);
    wmma_gemm<__bf16, false, true, true><<<gProj, 256, 0, stream>>>(
        Xb, Wvb, bv, vb, D_MODEL, D_MODEL, D_MODEL, D_MODEL,
        1, 0, 0, 0, 0, 0, 0, 1.0f);

    // -------- scores: per (b,h)  [2048,2048] = q_bh @ k_bh^T * 1/sqrt(64) --
    {
        dim3 g(SEQ / 64, SEQ / 256, BATCH * N_HEADS);
        wmma_gemm<__bf16, true, false, false><<<g, 256, 0, stream>>>(
            qb, kb, nullptr, attn, D_K,
            D_MODEL, D_MODEL, SEQ, N_HEADS,
            (long long)SEQ * D_MODEL, (long long)D_K,               // A: batch/head
            (long long)SEQ * D_MODEL, (long long)D_K,               // B: batch/head
            (long long)N_HEADS * SEQ * SEQ, (long long)SEQ * SEQ,   // C
            0.125f);
    }

    // -------- softmax over last dim, in place in d_out ---------------------
    softmax_rows2048<<<BATCH * N_HEADS * SEQ, 256, 0, stream>>>(attn);

    // -------- ctx: per (b,h)  [2048,64] = attn_bh @ v_bh  (f32 A on the fly)
    {
        dim3 g(D_K / 64, SEQ / 256, BATCH * N_HEADS);
        wmma_gemm<float, false, false, true><<<g, 256, 0, stream>>>(
            attn, vb, nullptr, ctx, SEQ,
            SEQ, D_MODEL, D_MODEL, N_HEADS,
            (long long)N_HEADS * SEQ * SEQ, (long long)SEQ * SEQ,   // A
            (long long)SEQ * D_MODEL, (long long)D_K,               // B
            (long long)SEQ * D_MODEL, (long long)D_K,               // C
            1.0f);
    }

    // -------- output projection: [4096,1024] = ctx @ Wo + bo (f32 out) -----
    wmma_gemm<__bf16, false, true, false><<<gProj, 256, 0, stream>>>(
        ctx, Wob, bo, out, D_MODEL, D_MODEL, D_MODEL, D_MODEL,
        1, 0, 0, 0, 0, 0, 0, 1.0f);
}
